// LSTMPolicy_25726854103279
// MI455X (gfx1250) — matrix operand order repaired and checked by reading, not compiled, not run on hardware
//
#include <hip/hip_runtime.h>
#include <math.h>

// Problem dims
#define OBS_DIM 128
#define MLP_HID 256
#define HID     256
#define ACT_DIM 8
#define TB      8192          // T*B flattened sequence length
#define GATES   1024          // 4*HID

typedef __attribute__((ext_vector_type(16))) _Float16 v16h;
typedef __attribute__((ext_vector_type(8)))  _Float16 v8h;
typedef __attribute__((ext_vector_type(8)))  float    v8f;
typedef __attribute__((ext_vector_type(4)))  float    v4f;

// ---------------------------------------------------------------------------
// Weight preparation: f32 -> f16, transposed into WMMA-B-friendly layout
// Bt[n*K + k] = W[k*N + n]  (lane n reads contiguous K)
// ---------------------------------------------------------------------------
__global__ void k_transpose_f16(const float* __restrict__ W,
                                _Float16* __restrict__ out, int K, int N) {
  int i = blockIdx.x * blockDim.x + threadIdx.x;
  if (i < K * N) {
    int n = i / K, k = i - n * K;
    out[i] = (_Float16)W[k * N + n];
  }
}

__global__ void k_convert_f16(const float* __restrict__ W,
                              _Float16* __restrict__ out, int n) {
  int i = blockIdx.x * blockDim.x + threadIdx.x;
  if (i < n) out[i] = (_Float16)W[i];
}

__global__ void k_bias_sum(const float* __restrict__ a,
                           const float* __restrict__ b,
                           float* __restrict__ out, int n) {
  int i = blockIdx.x * blockDim.x + threadIdx.x;
  if (i < n) out[i] = a[i] + b[i];
}

// ---------------------------------------------------------------------------
// WMMA GEMM: C[M,N] = epilogue(A[M,K] @ Bt^T + bias)
//   A: fp32 (AMODE=0) or f16 (AMODE=1), row-major [M,K]
//   Bt: f16, layout Bt[n*K + k]  (pre-transposed weights)
//   EPI: 0 = relu -> f16 out, 1 = f16 out, 2 = f32 out
// One wave computes a 16x64 output strip: ONE A fragment is reused across
// FOUR B fragments / accumulators per 32-wide K chunk (4x lower A traffic,
// 4 v_wmma per chunk). Fragment layouts follow CDNA5 ISA.
// ---------------------------------------------------------------------------
template<int AMODE, int EPI>
__global__ void __launch_bounds__(256)
k_gemm_wmma(const void* __restrict__ Av, const _Float16* __restrict__ Bt,
            const float* __restrict__ bias, void* __restrict__ Cv,
            int M, int N, int K) {
  const int wave    = blockIdx.x * (blockDim.x >> 5) + (threadIdx.x >> 5);
  const int tilesN4 = N >> 6;                 // 64-wide column strips
  const int wm      = wave / tilesN4;
  const int tm      = wm << 4;
  const int tn      = (wave - wm * tilesN4) << 6;
  const int lane    = threadIdx.x & 31;
  const int half_id = lane >> 4;
  const int ls      = lane & 15;

  const int rowA = tm + ls;   // A fragment: this lane's M row

  v8f acc[4] = {v8f{}, v8f{}, v8f{}, v8f{}};
  for (int kb = 0; kb < K; kb += 32) {
    v16h a;
    // ---- A fragment (lane<16: K {0..7,16..23}; lane>=16: {8..15,24..31}) ----
    const int ka = kb + half_id * 8;
    if (AMODE == 0) {
      const float* A = (const float*)Av;
      const v4f* p0 = (const v4f*)(A + (size_t)rowA * K + ka);
      const v4f* p1 = (const v4f*)(A + (size_t)rowA * K + ka + 16);
      v4f x0 = p0[0], x1 = p0[1], y0 = p1[0], y1 = p1[1];
#pragma unroll
      for (int j = 0; j < 4; ++j) {
        a[j]      = (_Float16)x0[j];
        a[4 + j]  = (_Float16)x1[j];
        a[8 + j]  = (_Float16)y0[j];
        a[12 + j] = (_Float16)y1[j];
      }
    } else {
      const _Float16* A = (const _Float16*)Av;
      v8h x = *(const v8h*)(A + (size_t)rowA * K + ka);
      v8h y = *(const v8h*)(A + (size_t)rowA * K + ka + 16);
#pragma unroll
      for (int j = 0; j < 8; ++j) { a[j] = x[j]; a[8 + j] = y[j]; }
    }
    // ---- 4 B fragments share the A fragment ----
    const int kcol = kb + half_id * 8;
#pragma unroll
    for (int s = 0; s < 4; ++s) {
      const int colB = tn + s * 16 + ls;      // this lane's N column
      v16h b;
      v8h b0 = *(const v8h*)(Bt + (size_t)colB * K + kcol);
      v8h b1 = *(const v8h*)(Bt + (size_t)colB * K + kcol + 16);
#pragma unroll
      for (int j = 0; j < 8; ++j) { b[j] = b0[j]; b[8 + j] = b1[j]; }
      acc[s] = __builtin_amdgcn_wmma_f32_16x16x32_f16(
          false, a, false, b, (short)0, acc[s], false, false);
    }
  }

  // ---- epilogue + store (C layout: element r -> row tm + half_id*8 + r) ----
#pragma unroll
  for (int s = 0; s < 4; ++s) {
#pragma unroll
    for (int r = 0; r < 8; ++r) {
      const int row = tm + half_id * 8 + r;
      const int col = tn + s * 16 + ls;
      float v = acc[s][r] + bias[col];
      if (EPI == 0) v = fmaxf(v, 0.0f);
      if (EPI <= 1) ((_Float16*)Cv)[(size_t)row * N + col] = (_Float16)v;
      else          ((float*)Cv)[(size_t)row * N + col]    = v;
    }
  }
}

// ---------------------------------------------------------------------------
// Sequential LSTM: single persistent workgroup (one WGP), 1024 threads.
// Thread j owns gate row j. W_hh row j is REGISTER-RESIDENT as 128 packed-f16
// dwords (512 KB of VGPRs workgroup-wide -> zero per-step weight traffic).
// The per-step pre_in row (4 KB) is double-buffered in LDS via
// global_load_async_to_lds_b128 (waves 0-7), overlapped with the matvec and
// fenced with s_wait_asynccnt. h lives in LDS; c lives in the owning thread.
// ---------------------------------------------------------------------------
__device__ __forceinline__ unsigned lds_off_of(const void* p) {
  // Generic pointers to LDS carry the wave-relative LDS offset in addr[31:0]
  // (CDNA5 aperture rule) -- exactly what the async-load VDST operand needs.
  return (unsigned)(unsigned long long)p;
}

__global__ void __launch_bounds__(1024, 1)
k_lstm(const float* __restrict__ W_hh, const float* __restrict__ pre_in,
       float* __restrict__ hs) {
  __shared__ __align__(16) float sh[HID];          // hidden state h
  __shared__ __align__(16) float sg[GATES];        // gate pre-activations
  __shared__ __align__(16) float spre[2][GATES];   // double-buffered pre_in row
  const int t = threadIdx.x;

  // Preload W_hh row t as packed f16 pairs into registers.
  unsigned wreg[HID / 2];
  {
    const float2* wrow = (const float2*)(W_hh + (size_t)t * HID);
#pragma unroll
    for (int v = 0; v < HID / 2; ++v) {
      float2 w2 = wrow[v];
      union { unsigned u; _Float16 h[2]; } p;
      p.h[0] = (_Float16)w2.x;
      p.h[1] = (_Float16)w2.y;
      wreg[v] = p.u;
    }
  }

  float c = 0.0f;
  if (t < HID) sh[t] = 0.0f;

  // Prime buffer 0 with the step-0 pre_in row (async copy to LDS).
  if (t < GATES / 4) {
    unsigned      dst = lds_off_of(&spre[0][t * 4]);
    unsigned long src = (unsigned long)(pre_in + (size_t)t * 4);
    asm volatile("global_load_async_to_lds_b128 %0, %1, off"
                 :: "v"(dst), "v"(src) : "memory");
  }
  asm volatile("s_wait_asynccnt 0" ::: "memory");
  __syncthreads();

  for (int step = 0; step < TB; ++step) {
    const int cur = step & 1;
    // Kick off async fetch of the NEXT row while we compute this step.
    if ((step + 1 < TB) && (t < GATES / 4)) {
      unsigned      dst = lds_off_of(&spre[cur ^ 1][t * 4]);
      unsigned long src = (unsigned long)(pre_in + (size_t)(step + 1) * GATES + t * 4);
      asm volatile("global_load_async_to_lds_b128 %0, %1, off"
                   :: "v"(dst), "v"(src) : "memory");
    }

    // gates[t] = pre_in[step][t] + dot(W_hh[t,:], h)
    float acc = spre[cur][t];
#pragma unroll
    for (int v = 0; v < HID / 4; ++v) {
      float4 h4 = ((const float4*)sh)[v];
      union { unsigned u; _Float16 h[2]; } p0, p1;
      p0.u = wreg[2 * v];
      p1.u = wreg[2 * v + 1];
      acc = fmaf((float)p0.h[0], h4.x, acc);
      acc = fmaf((float)p0.h[1], h4.y, acc);
      acc = fmaf((float)p1.h[0], h4.z, acc);
      acc = fmaf((float)p1.h[1], h4.w, acc);
    }
    sg[t] = acc;
    __syncthreads();

    if (t < HID) {
      float gi = sg[t], gf = sg[HID + t], gg = sg[2 * HID + t], go = sg[3 * HID + t];
      float i = 1.0f / (1.0f + __expf(-gi));
      float f = 1.0f / (1.0f + __expf(-gf));
      float g = tanhf(gg);
      float o = 1.0f / (1.0f + __expf(-go));
      c = f * c + i * g;
      float h = o * tanhf(c);
      sh[t] = h;
      hs[(size_t)step * HID + t] = h;
    }
    // Ensure next-row async copy landed, then release everyone into step+1.
    asm volatile("s_wait_asynccnt 0" ::: "memory");
    __syncthreads();
  }
}

// ---------------------------------------------------------------------------
// Head + NLL: one thread per (row, action) element; deterministic two-stage
// reduction (block LDS tree -> partials -> final single-block sum).
// ---------------------------------------------------------------------------
__global__ void __launch_bounds__(256)
k_head_loss(const float* __restrict__ hs, const float* __restrict__ act,
            const float* __restrict__ W_head, const float* __restrict__ b_head,
            float* __restrict__ partials) {
  const int idx = blockIdx.x * blockDim.x + threadIdx.x;  // < TB*ACT_DIM
  const int row = idx >> 3;
  const int a   = idx & 7;

  float accd = b_head[a];
  const float* h = hs + (size_t)row * HID;
#pragma unroll 8
  for (int k = 0; k < HID; ++k)
    accd = fmaf(h[k], W_head[k * ACT_DIM + a], accd);

  const float mu = tanhf(accd);
  const float z  = (act[idx] - mu) * 10.0f;               // / STD (0.1)
  // -log_prob = 0.5 z^2 + log(STD) + 0.5*log(2*pi)
  const float nll = 0.5f * z * z + (-2.3025850929940457f) + 0.5f * 1.8378770664093453f;

  __shared__ float red[256];
  red[threadIdx.x] = nll;
  __syncthreads();
#pragma unroll
  for (int s = 128; s > 0; s >>= 1) {
    if (threadIdx.x < s) red[threadIdx.x] += red[threadIdx.x + s];
    __syncthreads();
  }
  if (threadIdx.x == 0) partials[blockIdx.x] = red[0];
}

__global__ void __launch_bounds__(256)
k_final(const float* __restrict__ partials, float* __restrict__ out) {
  __shared__ float red[256];
  red[threadIdx.x] = partials[threadIdx.x];
  __syncthreads();
#pragma unroll
  for (int s = 128; s > 0; s >>= 1) {
    if (threadIdx.x < s) red[threadIdx.x] += red[threadIdx.x + s];
    __syncthreads();
  }
  if (threadIdx.x == 0) out[0] = red[0] * (1.0f / (float)(TB * ACT_DIM));
}

// ---------------------------------------------------------------------------
// Launch
// ---------------------------------------------------------------------------
extern "C" void kernel_launch(void* const* d_in, const int* in_sizes, int n_in,
                              void* d_out, int out_size, void* d_ws, size_t ws_size,
                              hipStream_t stream) {
  (void)in_sizes; (void)n_in; (void)out_size; (void)ws_size;

  const float* obs    = (const float*)d_in[0];   // [TB, OBS_DIM]
  const float* act    = (const float*)d_in[1];   // [TB, ACT_DIM]
  const float* W1     = (const float*)d_in[2];   // [128, 256]
  const float* b1     = (const float*)d_in[3];
  const float* W2     = (const float*)d_in[4];   // [256, 256]
  const float* b2     = (const float*)d_in[5];
  const float* W_ih   = (const float*)d_in[6];   // [1024, 256]
  const float* W_hh   = (const float*)d_in[7];   // [1024, 256]
  const float* b_ih   = (const float*)d_in[8];
  const float* b_hh   = (const float*)d_in[9];
  const float* W_head = (const float*)d_in[10];  // [256, 8]
  const float* b_head = (const float*)d_in[11];
  float* out = (float*)d_out;

  // Workspace carve-up (256B aligned)
  char* ws = (char*)d_ws;
  size_t off = 0;
  auto alloc = [&](size_t bytes) { void* p = ws + off; off = (off + bytes + 255) & ~(size_t)255; return p; };
  _Float16* W1t     = (_Float16*)alloc((size_t)OBS_DIM * MLP_HID * 2);
  _Float16* W2t     = (_Float16*)alloc((size_t)MLP_HID * HID * 2);
  _Float16* Wiht    = (_Float16*)alloc((size_t)GATES * HID * 2);    // W_ih.T layout == W_ih rows
  float*    bsum    = (float*)   alloc((size_t)GATES * 4);
  _Float16* h1      = (_Float16*)alloc((size_t)TB * MLP_HID * 2);
  _Float16* emb     = (_Float16*)alloc((size_t)TB * HID * 2);
  float*    pre_in  = (float*)   alloc((size_t)TB * GATES * 4);
  float*    hs      = (float*)   alloc((size_t)TB * HID * 4);
  float*    partial = (float*)   alloc(256 * 4);

  // 1) Weight prep
  k_transpose_f16<<<(OBS_DIM * MLP_HID + 255) / 256, 256, 0, stream>>>(W1, W1t, OBS_DIM, MLP_HID);
  k_transpose_f16<<<(MLP_HID * HID + 255) / 256, 256, 0, stream>>>(W2, W2t, MLP_HID, HID);
  k_convert_f16<<<(GATES * HID + 255) / 256, 256, 0, stream>>>(W_ih, Wiht, GATES * HID);
  k_bias_sum<<<(GATES + 255) / 256, 256, 0, stream>>>(b_ih, b_hh, bsum, GATES);

  // 2) Parallel GEMMs (WMMA f16, f32 accumulate); 16x64 strip per wave
  //    h1 = relu(obs @ W1 + b1)     : [8192,256], K=128
  k_gemm_wmma<0, 0><<<(TB / 16) * (MLP_HID / 64) / 8, 256, 0, stream>>>(
      obs, W1t, b1, h1, TB, MLP_HID, OBS_DIM);
  //    emb = h1 @ W2 + b2           : [8192,256], K=256
  k_gemm_wmma<1, 1><<<(TB / 16) * (HID / 64) / 8, 256, 0, stream>>>(
      h1, W2t, b2, emb, TB, HID, MLP_HID);
  //    pre_in = emb @ W_ih.T + (b_ih+b_hh) : [8192,1024], K=256
  k_gemm_wmma<1, 2><<<(TB / 16) * (GATES / 64) / 8, 256, 0, stream>>>(
      emb, Wiht, bsum, pre_in, TB, GATES, HID);

  // 3) Sequential LSTM recurrence (single WGP, register-resident W_hh,
  //    async-to-LDS double-buffered pre_in)
  k_lstm<<<1, 1024, 0, stream>>>(W_hh, pre_in, hs);

  // 4) Head + NLL + deterministic reduction
  k_head_loss<<<256, 256, 0, stream>>>(hs, act, W_head, b_head, partial);
  k_final<<<1, 256, 0, stream>>>(partial, out);
}
